// MultiheadAttention_47399259079138
// MI455X (gfx1250) — compile-verified, matrix operand order checked by reading, not run on hardware
//
#include <hip/hip_runtime.h>
#include <hip/hip_bf16.h>

// ---------------------------------------------------------------------------
// MultiheadAttention forward, CDNA5 (gfx1250) WMMA implementation.
// B=2, S=2048, D=1024, H=16, DK=64. All GEMMs run on v_wmma_f32_16x16x32_f16.
// ---------------------------------------------------------------------------

#define BB 2
#define SS 2048
#define DD 1024
#define HH 16
#define DKK 64

typedef __attribute__((ext_vector_type(16))) _Float16 v16h;
typedef __attribute__((ext_vector_type(8)))  _Float16 v8h;
typedef __attribute__((ext_vector_type(4)))  _Float16 v4h;
typedef __attribute__((ext_vector_type(8)))  float    v8f;
typedef __attribute__((ext_vector_type(4)))  float    v4f;

// ---- WMMA helpers ----------------------------------------------------------
// A/B 16-bit operand layout (ISA 7.12.2): lane L holds row (A) / column (B)
// index (L & 15); halves 0..7 are K = kbase..kbase+7 with kbase = k + (L>>4)*8,
// halves 8..15 are K = kbase+16..kbase+23.  Both windows are 16B contiguous.
__device__ __forceinline__ v16h load_ab16(const _Float16* __restrict__ p) {
  v8h lo = *(const v8h*)(p);
  v8h hi = *(const v8h*)(p + 16);
  v16h r;
#pragma unroll
  for (int i = 0; i < 8; ++i) { r[i] = lo[i]; r[i + 8] = hi[i]; }
  return r;
}

__device__ __forceinline__ v16h load_ab16_lds(const _Float16* p) {
  v8h lo = *(const v8h*)(p);
  v8h hi = *(const v8h*)(p + 16);
  v16h r;
#pragma unroll
  for (int i = 0; i < 8; ++i) { r[i] = lo[i]; r[i + 8] = hi[i]; }
  return r;
}

__device__ __forceinline__ v8f wmma16(v16h a, v16h b, v8f c) {
  return __builtin_amdgcn_wmma_f32_16x16x32_f16(
      /*neg_a=*/false, a, /*neg_b=*/false, b,
      /*c_mod=*/(short)0, c, /*reuse_a=*/false, /*reuse_b=*/false);
}

// Wave-local LDS ordering: per-wave LDS ops are hardware-in-order (ISA 7.3),
// so a compiler-level fence is all that's needed for the P-tile bounce --
// no s_barrier, waves in the block never rendezvous.
__device__ __forceinline__ void wave_lds_fence() {
  __builtin_amdgcn_fence(__ATOMIC_ACQ_REL, "wavefront");
}

// ---- f32 -> f16 conversion (vectorized x4) ---------------------------------
__global__ __launch_bounds__(256) void cvt_f32_f16(const float* __restrict__ src,
                                                   _Float16* __restrict__ dst,
                                                   int n4) {
  int i = blockIdx.x * blockDim.x + threadIdx.x;
  if (i < n4) {
    v4f x = ((const v4f*)src)[i];
    v4h y;
#pragma unroll
    for (int j = 0; j < 4; ++j) y[j] = (_Float16)x[j];
    ((v4h*)dst)[i] = y;
  }
}

// ---- Tiled GEMM: C[M,N] = A[M,K] @ Bt[N,K]^T + bias ------------------------
// Each wave owns a 32x64 output strip: 2 A row-tiles reuse every B tile load
// (8 WMMA per 12 b128 loads per K-step). MODE 0: scatter to headed [B,H,S,DK]
// f16. MODE 1: v-transposed [B,H,DK,S] f16 -- C-layout i maps to contiguous s,
// so each lane stores one packed v8h (b128) per (r,j). MODE 2: f32 row-major.
template <int MODE>
__global__ __launch_bounds__(256) void gemm_ws(const _Float16* __restrict__ A,
                                               const _Float16* __restrict__ Bt,
                                               const float* __restrict__ bias,
                                               void* __restrict__ Cout,
                                               int M, int N, int K) {
  const int lane = threadIdx.x & 31;
  const int wave = threadIdx.x >> 5;
  const int col  = lane & 15;
  const int hh   = lane >> 4;          // half-wave select (K / row-group split)
  const int ntc  = N >> 6;             // column strips of 64
  const int wid  = blockIdx.x * 8 + wave;
  const int tr   = (wid / ntc) << 5;   // tile row base (32 rows)
  const int tc   = (wid % ntc) << 6;   // tile col base (64 cols)
  if (tr >= M) return;

  const _Float16* arow0 = A + (size_t)(tr + col) * K;
  const _Float16* arow1 = A + (size_t)(tr + 16 + col) * K;
  v8f acc[2][4] = {};

#pragma unroll 2
  for (int k = 0; k < K; k += 32) {
    const int koff = k + hh * 8;
    __builtin_prefetch(arow0 + koff + 64, 0, 1);
    __builtin_prefetch(arow1 + koff + 64, 0, 1);
    v16h a0 = load_ab16(arow0 + koff);
    v16h a1 = load_ab16(arow1 + koff);
#pragma unroll
    for (int j = 0; j < 4; ++j) {
      v16h b = load_ab16(Bt + (size_t)(tc + j * 16 + col) * K + koff);
      acc[0][j] = wmma16(a0, b, acc[0][j]);
      acc[1][j] = wmma16(a1, b, acc[1][j]);
    }
  }

  // C layout: lane covers column n = tc + (lane&15); VGPR i covers row
  // m = tilerow + (lane>>4)*8 + i.
#pragma unroll
  for (int r = 0; r < 2; ++r) {
#pragma unroll
    for (int j = 0; j < 4; ++j) {
      const int n = tc + j * 16 + col;
      const float bz = bias[n];
      if (MODE == 1) {
        // 8 accumulator rows = 8 contiguous s in [B,H,DK,S]: one b128 store.
        const int m0 = tr + r * 16 + hh * 8;
        const int b = m0 / SS, s = m0 % SS;
        const int h = n / DKK, dk = n % DKK;
        v8h pk;
#pragma unroll
        for (int i = 0; i < 8; ++i) pk[i] = (_Float16)(acc[r][j][i] + bz);
        *(v8h*)&((_Float16*)Cout)[(((size_t)b * HH + h) * DKK + dk) * SS + s] = pk;
      } else {
#pragma unroll
        for (int i = 0; i < 8; ++i) {
          const int m = tr + r * 16 + hh * 8 + i;
          const float v = acc[r][j][i] + bz;
          if (MODE == 0) {          // headed: [b][h][s][dk] f16
            const int b = m / SS, s = m % SS;
            const int h = n / DKK, dk = n % DKK;
            ((_Float16*)Cout)[(((size_t)b * HH + h) * SS + s) * DKK + dk] = (_Float16)v;
          } else {                  // plain f32 row-major
            ((float*)Cout)[(size_t)m * N + n] = v;
          }
        }
      }
    }
  }
}

// ---- Flash attention -------------------------------------------------------
// Per-wave 32-row q tile; 64 keys per softmax round. q is pre-scaled by
// 1/sqrt(DK) * log2(e) so the softmax runs in the exp2 domain with no extra
// multiplies. Row sums come from a WMMA against an all-ones B tile (flash
// rescaling applies identically), so only the row-max needs lane butterflies.
__global__ __launch_bounds__(256) void attn_fwd(const _Float16* __restrict__ q,
                                                const _Float16* __restrict__ k,
                                                const _Float16* __restrict__ vT,
                                                _Float16* __restrict__ ctx) {
  __shared__ __align__(16) _Float16 pbuf[8][32 * 64];   // per-wave P tile (4KB)
  const int lane = threadIdx.x & 31;
  const int wave = threadIdx.x >> 5;
  const int col  = lane & 15;
  const int hh   = lane >> 4;
  const int wid  = blockIdx.x * 8 + wave;  // 0 .. B*H*(S/32)-1
  const int qt   = wid % (SS / 32);
  const int bh   = wid / (SS / 32);

  const _Float16* qbase = q  + (size_t)bh * SS * DKK;
  const _Float16* kbase = k  + (size_t)bh * SS * DKK;
  const _Float16* vbase = vT + (size_t)bh * DKK * SS;

  // q tiles (2 row groups x 2 K-chunks), pre-scaled: 0.125 * log2(e)
  const _Float16 hs = (_Float16)0.18033688f;
  v16h qa[2][2];
#pragma unroll
  for (int r = 0; r < 2; ++r) {
    const _Float16* qrow = qbase + (size_t)(qt * 32 + r * 16 + col) * DKK;
    qa[r][0] = load_ab16(qrow + hh * 8) * hs;
    qa[r][1] = load_ab16(qrow + 32 + hh * 8) * hs;
  }

  v16h ones;
#pragma unroll
  for (int i = 0; i < 16; ++i) ones[i] = (_Float16)1.0f;

  v8f acc[2][4] = {};
  v8f accS[2] = {};
  float mrow[2][8];
#pragma unroll
  for (int r = 0; r < 2; ++r)
#pragma unroll
    for (int i = 0; i < 8; ++i) mrow[r][i] = -1e30f;

  _Float16* pb = pbuf[wave];

  for (int kt = 0; kt < SS; kt += 64) {
    // ---- scores: 4 key tiles x 2 row groups; K/V b128s reused across rows --
    v8f s[2][4];
#pragma unroll
    for (int t = 0; t < 4; ++t) {
      const _Float16* kr = kbase + (size_t)(kt + t * 16 + col) * DKK;
      v16h kb0 = load_ab16(kr + hh * 8);
      v16h kb1 = load_ab16(kr + 32 + hh * 8);
      v8f z = {};
      s[0][t] = wmma16(qa[0][1], kb1, wmma16(qa[0][0], kb0, z));
      s[1][t] = wmma16(qa[1][1], kb1, wmma16(qa[1][0], kb0, z));
    }
    // ---- online softmax (exp2 domain); one max butterfly per 64 keys ------
#pragma unroll
    for (int r = 0; r < 2; ++r) {
#pragma unroll
      for (int i = 0; i < 8; ++i) {
        float mx = fmaxf(fmaxf(s[r][0][i], s[r][1][i]),
                         fmaxf(s[r][2][i], s[r][3][i]));
#pragma unroll
        for (int d = 1; d < 16; d <<= 1) mx = fmaxf(mx, __shfl_xor(mx, d, 32));
        const float mnew = fmaxf(mrow[r][i], mx);
        const float corr = exp2f(mrow[r][i] - mnew);
        mrow[r][i] = mnew;
#pragma unroll
        for (int j = 0; j < 4; ++j) acc[r][j][i] *= corr;
        accS[r][i] *= corr;
        const int row = r * 16 + hh * 8 + i;
#pragma unroll
        for (int t = 0; t < 4; ++t) {
          pb[row * 64 + t * 16 + col] = (_Float16)exp2f(s[r][t][i] - mnew);
        }
      }
    }
    wave_lds_fence();
    // ---- P back in A layout (two 16x32 chunks per row group) --------------
    v16h pa[2][2];
#pragma unroll
    for (int r = 0; r < 2; ++r) {
      const _Float16* prow = pb + (r * 16 + col) * 64;
      pa[r][0] = load_ab16_lds(prow + hh * 8);
      pa[r][1] = load_ab16_lds(prow + 32 + hh * 8);
    }
    wave_lds_fence();
    // ---- row sums: fold the two K-chunks with packed f16 adds, then one
    //      ones-WMMA per row group; ctx += P @ V (vT rows are contiguous) ----
#pragma unroll
    for (int r = 0; r < 2; ++r)
      accS[r] = wmma16(pa[r][0] + pa[r][1], ones, accS[r]);
#pragma unroll
    for (int j = 0; j < 4; ++j) {
      const _Float16* vr = vbase + (size_t)(j * 16 + col) * SS + kt;
      v16h vb0 = load_ab16(vr + hh * 8);
      v16h vb1 = load_ab16(vr + 32 + hh * 8);
#pragma unroll
      for (int r = 0; r < 2; ++r)
        acc[r][j] = wmma16(pa[r][1], vb1, wmma16(pa[r][0], vb0, acc[r][j]));
    }
  }

  // ---- finalize: divide by row sums, store [b][s][h*64+dk] f16 ----
  const int b = bh / HH, h = bh % HH;
#pragma unroll
  for (int r = 0; r < 2; ++r) {
#pragma unroll
    for (int i = 0; i < 8; ++i) {
      const float inv = 1.0f / accS[r][i];
      const int s = qt * 32 + r * 16 + hh * 8 + i;
#pragma unroll
      for (int j = 0; j < 4; ++j) {
        const int dk = j * 16 + col;
        ctx[((size_t)(b * SS + s)) * DD + h * DKK + dk] =
            (_Float16)(acc[r][j][i] * inv);
      }
    }
  }
}

// ---------------------------------------------------------------------------
extern "C" void kernel_launch(void* const* d_in, const int* in_sizes, int n_in,
                              void* d_out, int out_size, void* d_ws, size_t ws_size,
                              hipStream_t stream) {
  const float* Q  = (const float*)d_in[0];
  const float* K  = (const float*)d_in[1];
  const float* V  = (const float*)d_in[2];
  const float* Wq = (const float*)d_in[3];
  const float* bq = (const float*)d_in[4];
  const float* Wk = (const float*)d_in[5];
  const float* bk = (const float*)d_in[6];
  const float* Wv = (const float*)d_in[7];
  const float* bv = (const float*)d_in[8];
  const float* Wo = (const float*)d_in[9];
  const float* bo = (const float*)d_in[10];
  float* out = (float*)d_out;

  const size_t XN = (size_t)BB * SS * DD;   // 4M elems
  const size_t WN = (size_t)DD * DD;        // 1M elems

  _Float16* w = (_Float16*)d_ws;
  _Float16* Qh   = w;                 // 4M
  _Float16* Kh   = Qh  + XN;          // 4M
  _Float16* Vh   = Kh  + XN;          // 4M
  _Float16* Wqh  = Vh  + XN;          // 1M
  _Float16* Wkh  = Wqh + WN;          // 1M
  _Float16* Wvh  = Wkh + WN;          // 1M
  _Float16* Woh  = Wvh + WN;          // 1M
  _Float16* qh   = Woh + WN;          // 4M  [B,H,S,DK]
  _Float16* kh   = qh  + XN;          // 4M  [B,H,S,DK]
  _Float16* vTh  = kh  + XN;          // 4M  [B,H,DK,S]
  _Float16* ctxh = vTh + XN;          // 4M  [B*S, D]

  // 1) convert everything that feeds a WMMA to f16
  {
    const int t = 256;
    int n4x = (int)(XN / 4), n4w = (int)(WN / 4);
    cvt_f32_f16<<<(n4x + t - 1) / t, t, 0, stream>>>(Q,  Qh,  n4x);
    cvt_f32_f16<<<(n4x + t - 1) / t, t, 0, stream>>>(K,  Kh,  n4x);
    cvt_f32_f16<<<(n4x + t - 1) / t, t, 0, stream>>>(V,  Vh,  n4x);
    cvt_f32_f16<<<(n4w + t - 1) / t, t, 0, stream>>>(Wq, Wqh, n4w);
    cvt_f32_f16<<<(n4w + t - 1) / t, t, 0, stream>>>(Wk, Wkh, n4w);
    cvt_f32_f16<<<(n4w + t - 1) / t, t, 0, stream>>>(Wv, Wvh, n4w);
    cvt_f32_f16<<<(n4w + t - 1) / t, t, 0, stream>>>(Wo, Woh, n4w);
  }

  // 2) projections: M=B*S=4096, N=D=1024, K=D=1024; 32x64 strip per wave
  {
    const int M = BB * SS, N = DD, Kd = DD;
    const int blocks = (M / 32) * (N / 64) / 8;   // 256
    gemm_ws<0><<<blocks, 256, 0, stream>>>(Qh, Wqh, bq, qh,  M, N, Kd);
    gemm_ws<0><<<blocks, 256, 0, stream>>>(Kh, Wkh, bk, kh,  M, N, Kd);
    gemm_ws<1><<<blocks, 256, 0, stream>>>(Vh, Wvh, bv, vTh, M, N, Kd);
  }

  // 3) flash attention: B*H*(S/32) = 2048 waves -> 256 blocks of 8 waves
  attn_fwd<<<(BB * HH * (SS / 32)) / 8, 256, 0, stream>>>(qh, kh, vTh, ctxh);

  // 4) output projection: f32 out + bias
  {
    const int M = BB * SS, N = DD, Kd = DD;
    const int blocks = (M / 32) * (N / 64) / 8;
    gemm_ws<2><<<blocks, 256, 0, stream>>>(ctxh, Woh, bo, out, M, N, Kd);
  }
}
